// DispEstimator_22840636080139
// MI455X (gfx1250) — matrix operand order repaired and checked by reading, not compiled, not run on hardware
//
#include <hip/hip_runtime.h>
#include <hip/hip_bf16.h>

// ---------- types for CDNA5 WMMA ----------
typedef __attribute__((ext_vector_type(16))) __bf16 v16bf;
typedef __attribute__((ext_vector_type(8)))  float  v8f;
typedef __attribute__((ext_vector_type(8)))  unsigned short v8u16;

// ---------- bf16 helpers (bit-level, round-to-nearest-even) ----------
__device__ __forceinline__ unsigned short f2bf_bits(float f) {
    union { float f; unsigned u; } v; v.f = f;
    unsigned r = v.u + 0x7FFFu + ((v.u >> 16) & 1u);
    return (unsigned short)(r >> 16);
}
__device__ __forceinline__ float bf2f(unsigned short h) {
    union { unsigned u; float f; } v; v.u = ((unsigned)h) << 16;
    return v.f;
}

// ================= elementwise f32 -> bf16 convert =================
__global__ void cvt_f32_bf16(const float* __restrict__ src,
                             unsigned short* __restrict__ dst, size_t n) {
    size_t i = (size_t)blockIdx.x * blockDim.x + threadIdx.x;
    if (i < n) dst[i] = f2bf_bits(src[i]);
}

// ================= weight packing into A-fragment order ============
// dst layout: [group = oc/16][chunk = K/32][lane 0..31][e 0..15]
// A fragment mapping used by this code (contiguous k, matched by B from LDS):
//   M = lane%16 ; K = e + 16*(lane/16)
// K ordering for the implicit GEMM: K = r*CinPad + ci, r = ky*3+kx
__global__ void pack_w(const float* __restrict__ w, unsigned short* __restrict__ dst,
                       int Cout, int CinSrc, int CinPad) {
    int chunks = (9 * CinPad) >> 5;
    int total  = (Cout >> 4) * chunks * 512;
    int idx = blockIdx.x * blockDim.x + threadIdx.x;
    if (idx >= total) return;
    int e     = idx & 15;
    int lane  = (idx >> 4) & 31;
    int chunk = (idx >> 9) % chunks;
    int group = idx / (chunks * 512);
    int lh    = lane >> 4;
    int kidx  = e + 16 * lh;
    int K     = chunk * 32 + kidx;
    int r     = K / CinPad;
    int ci    = K % CinPad;
    int oc    = group * 16 + (lane & 15);
    float v = 0.f;
    if (ci < CinSrc) v = w[((size_t)oc * CinSrc + ci) * 9 + r];
    dst[idx] = f2bf_bits(v);
}

// ================= implicit-GEMM 3x3 conv via WMMA bf16 ============
// Block = 64 threads (2 waves) -> one 16-pixel row segment, ALL Cout groups.
// Stage: global NCHW bf16 -> LDS [row 0..2][x 0..31][ci] (ci innermost) so a
// B fragment (16 consecutive channels at one pixel) is ONE aligned 32B LDS
// read. K-chunk OUTER / oc-group INNER register blocking: each chunk's B
// fragment is read from LDS once and immediately consumed by NG WMMAs
// (NG = Cout/32 groups per wave) -> no loop-invariant hoisting, no spills,
// bounded VGPR use (NG*8 accumulators + 8 B + transient A).
template <int CIN, int COUT, bool LRELU, bool OUT_BF16>
__global__ void conv3x3_wmma(const unsigned short* __restrict__ in0,
                             const unsigned short* __restrict__ in1,
                             int csplit,
                             const unsigned short* __restrict__ wpack,
                             float* __restrict__ outF,
                             unsigned short* __restrict__ outB) {
    constexpr int H = 256, W = 256;
    constexpr int KCH = (CIN * 9) / 32;   // total K chunks of 32
    constexpr int CC  = CIN / 32;         // chunks per filter tap
    constexpr int SX  = 32;               // staged pixels per row (aligned)
    constexpr int NG  = COUT / 32;        // oc groups per wave
    __shared__ unsigned short lb[3 * SX * CIN];

    const int tid  = threadIdx.x;         // 0..63
    const int lane = tid & 31;
    const int wave = tid >> 5;
    const int tile = blockIdx.x;          // pixel tiles (all batches)
    const int n  = tile >> 12;            // / (HW/16)
    const int p  = (tile & 4095) << 4;    // base pixel
    const int y  = p >> 8;
    const int xb = p & 255;
    const int xs = xb - 8;                // aligned staging window start

    // ---- cooperative stage: wide global b128 reads, ds transpose ----
    constexpr int NV = 3 * (SX / 8) * CIN;  // 8-pixel vectors to stage
    for (int v = tid; v < NV; v += 64) {
        const int seg = v & 3;              // SX/8 = 4 segments
        const int ry  = (v >> 2) % 3;
        const int ci  = (v >> 2) / 3;
        const int iy  = y + ry - 1;
        const int gx  = xs + (seg << 3);
        const unsigned short* src =
            (ci < csplit)
                ? (in0 + (((size_t)n * csplit + ci) << 16))
                : (in1 + (((size_t)n * (CIN - csplit) + (ci - csplit)) << 16));
        v8u16 t;
        const bool fast = (iy >= 0) & (iy < H) & (gx >= 0) & (gx + 8 <= W);
        if (fast) {
            t = *(const v8u16*)(src + (iy << 8) + gx);
        } else {
#pragma unroll
            for (int k = 0; k < 8; ++k) {
                int ix = gx + k;
                t[k] = (iy >= 0 && iy < H && ix >= 0 && ix < W)
                           ? src[(iy << 8) + ix] : (unsigned short)0;
            }
        }
        unsigned short* dst = &lb[((ry * SX + (seg << 3)) * CIN) + ci];
#pragma unroll
        for (int k = 0; k < 8; ++k) dst[k * CIN] = t[k];
    }
    __syncthreads();

    // ---- WMMA compute: chunk outer, oc-group inner ----
    const int lh = lane >> 4;
    const int ln = lane & 15;
    const unsigned short* bbase = &lb[(size_t)ln * CIN + lh * 16];

    // per-group A stream base pointers (groups wave, wave+2, ..., stride 2)
    const unsigned short* wa[NG];
#pragma unroll
    for (int gg = 0; gg < NG; ++gg)
        wa[gg] = wpack + (size_t)(wave + 2 * gg) * KCH * 512 + lane * 16;

    v8f acc[NG];
#pragma unroll
    for (int gg = 0; gg < NG; ++gg)
#pragma unroll
        for (int i = 0; i < 8; ++i) acc[gg][i] = 0.0f;

#pragma unroll
    for (int ky = 0; ky < 3; ++ky) {
#pragma unroll
        for (int kx = 0; kx < 3; ++kx) {
            const int tap = ky * 3 + kx;
            const unsigned short* bp = bbase + (ky * SX + kx + 7) * CIN;
#pragma unroll
            for (int cc = 0; cc < CC; ++cc) {
                const int chunk = tap * CC + cc;
                v16bf bfrag = *(const v16bf*)(bp + cc * 32);     // one 32B LDS read
#pragma unroll
                for (int gg = 0; gg < NG; ++gg) {
                    v16bf afrag = *(const v16bf*)(wa[gg] + chunk * 512);
                    acc[gg] = __builtin_amdgcn_wmma_f32_16x16x32_bf16(
                        false, afrag, false, bfrag, (short)0, acc[gg], false, false);
                }
            }
        }
    }

    // ---- store D: M = rr + 8*lh, N = ln ----
#pragma unroll
    for (int gg = 0; gg < NG; ++gg) {
        const int g = wave + 2 * gg;
#pragma unroll
        for (int rr = 0; rr < 8; ++rr) {
            int oc = g * 16 + rr + 8 * lh;
            float v = acc[gg][rr];
            if (LRELU) v = (v >= 0.f) ? v : 0.1f * v;
            size_t o = (((size_t)n * COUT + oc) << 16) + (size_t)(y << 8) + xb + ln;
            if (OUT_BF16) outB[o] = f2bf_bits(v);
            else          outF[o] = v;
        }
    }
}

// ================= 13-tap separable Gaussian blur ==================
__device__ __forceinline__ void gauss13(float* g) {
    float s = 0.f;
#pragma unroll
    for (int i = 0; i < 13; ++i) { float d = (float)i - 6.f; g[i] = __expf(-(d * d) / 18.f); s += g[i]; }
#pragma unroll
    for (int i = 0; i < 13; ++i) g[i] /= s;
}

// VERT: blur along y else x. EDGE: clamp pad else zero pad. INBF: bf16 input.
// FINAL: clip(+-300) then divide by (W-1 | H-1) per channel, f32 out.
template <bool VERT, bool EDGE, bool INBF, bool FINAL>
__global__ void blur13_pass(const void* __restrict__ src_, float* __restrict__ dst, int total) {
    int idx = blockIdx.x * blockDim.x + threadIdx.x;
    if (idx >= total) return;
    float g[13]; gauss13(g);
    const int p = idx & 65535, nc = idx >> 16;
    const int y = p >> 8, x = p & 255;
    float acc = 0.f;
#pragma unroll
    for (int t = 0; t < 13; ++t) {
        int yy = y, xx = x;
        if (VERT) yy = y + t - 6; else xx = x + t - 6;
        if (EDGE) { yy = min(max(yy, 0), 255); xx = min(max(xx, 0), 255); }
        else if (yy < 0 || yy >= 256 || xx < 0 || xx >= 256) continue;
        int off = (nc << 16) + (yy << 8) + xx;
        float v = INBF ? bf2f(((const unsigned short*)src_)[off])
                       : ((const float*)src_)[off];
        acc += g[t] * v;
    }
    if (FINAL) {
        acc = fminf(fmaxf(acc, -300.f), 300.f);
        acc *= (1.0f / 255.0f);             // (W-1) = (H-1) = 255
    }
    dst[idx] = acc;
}

// ============ assemble corr tensor: feat channels + zero pad =======
__global__ void feat_to_corr(const float* __restrict__ feat,
                             unsigned short* __restrict__ corr, int total) {
    int idx = blockIdx.x * blockDim.x + threadIdx.x;
    if (idx >= total) return;
    int c = (idx >> 16) & 127;
    if (c < 64) {
        int n = idx >> 23;                  // /(128*65536)
        int p = idx & 65535;
        corr[idx] = f2bf_bits(feat[(((size_t)n * 64 + c) << 16) + p]);
    } else if (c >= 113) {
        corr[idx] = 0;
    }
}

// ============ 49-shift local correlation ===========================
__global__ void localcorr_k(const unsigned short* __restrict__ f2bits,
                            const float* __restrict__ f1blur,
                            unsigned short* __restrict__ corr, int total) {
    int idx = blockIdx.x * blockDim.x + threadIdx.x;
    if (idx >= total) return;
    const int p = idx & 65535;
    const int s = (idx >> 16) % 49;
    const int n = idx / (49 << 16);
    const int y = p >> 8, x = p & 255;
    const int di = s / 7, dj = s % 7;
    const int sy = y + di * 4 - 12;
    const int sx = x + dj * 4 - 12;
    const bool inb = (sy >= 0) & (sy < 256) & (sx >= 0) & (sx < 256);
    const unsigned short* f2p = f2bits + (((size_t)n * 64) << 16) + p;
    const float* f1p = f1blur + (((size_t)n * 64) << 16) + (sy << 8) + sx;
    float acc = 0.f;
    for (int c = 0; c < 64; ++c) {
        float a = bf2f(f2p[c << 16]);
        float b = inb ? f1p[c << 16] : 0.f;
        float d = a - b;
        acc += d * d;
    }
    acc *= (1.f / 64.f);
    corr[(((size_t)n * 128 + 64 + s) << 16) + p] = f2bf_bits(acc);
}

// ================= BatchNorm (stats over N,H,W) ====================
__global__ void zero_f32(float* p, int n) {
    int i = blockIdx.x * blockDim.x + threadIdx.x;
    if (i < n) p[i] = 0.f;
}
__global__ void bn_stats(const float* __restrict__ x, float* __restrict__ stats,
                         int C, int total) {
    __shared__ float ssum[64];
    int t = threadIdx.x;
    if (t < 2 * C) ssum[t] = 0.f;
    __syncthreads();
    int idx = blockIdx.x * blockDim.x + t;
    if (idx < total) {
        float v = x[idx];
        int c = (idx >> 16) % C;            // HW = 65536
        atomicAdd(&ssum[c], v);
        atomicAdd(&ssum[C + c], v * v);
    }
    __syncthreads();
    if (t < 2 * C) atomicAdd(&stats[t], ssum[t]);
}
__global__ void bn_apply_lrelu(float* __restrict__ x, const float* __restrict__ stats,
                               int C, int total) {
    int idx = blockIdx.x * blockDim.x + threadIdx.x;
    if (idx >= total) return;
    int c = (idx >> 16) % C;
    float cnt = (float)(total / C);
    float m = stats[c] / cnt;
    float var = stats[C + c] / cnt - m * m;
    float v = (x[idx] - m) * rsqrtf(var + 1e-5f);
    x[idx] = (v >= 0.f) ? v : 0.1f * v;
}

// ================= small dilated direct conv (f32) =================
__global__ void conv3x3_direct(const float* __restrict__ in, const float* __restrict__ w,
                               float* __restrict__ out, int Cin, int Cout, int dil, int total) {
    int idx = blockIdx.x * blockDim.x + threadIdx.x;
    if (idx >= total) return;
    const int p  = idx & 65535;
    const int oc = (idx >> 16) % Cout;
    const int n  = idx / (Cout << 16);
    const int y = p >> 8, x = p & 255;
    float acc = 0.f;
    for (int ky = 0; ky < 3; ++ky) {
        int iy = y + (ky - 1) * dil;
        if (iy < 0 || iy >= 256) continue;
        for (int kx = 0; kx < 3; ++kx) {
            int ix = x + (kx - 1) * dil;
            if (ix < 0 || ix >= 256) continue;
            const float* ip = in + (((size_t)n * Cin) << 16) + (iy << 8) + ix;
            const float* wp = w + (size_t)oc * Cin * 9 + ky * 3 + kx;
            for (int ci = 0; ci < Cin; ++ci) acc += ip[(size_t)ci << 16] * wp[ci * 9];
        }
    }
    out[idx] = acc;
}

// ===================================================================
extern "C" void kernel_launch(void* const* d_in, const int* in_sizes, int n_in,
                              void* d_out, int out_size, void* d_ws, size_t ws_size,
                              hipStream_t stream) {
    (void)in_sizes; (void)n_in; (void)out_size;
    constexpr size_t HW = 65536;

    const float* feat1 = (const float*)d_in[0];
    const float* feat2 = (const float*)d_in[1];
    const float* w_pre = (const float*)d_in[2];
    const float* w_fc1 = (const float*)d_in[3];
    const float* w_fc2 = (const float*)d_in[4];
    const float* w_e0  = (const float*)d_in[5];
    const float* w_e1  = (const float*)d_in[6];
    const float* w_e2  = (const float*)d_in[7];
    const float* w_e3  = (const float*)d_in[8];

    // ---- workspace layout (bytes) ----
    char* ws = (char*)d_ws;
    const size_t OFF_R1 = 67108864, OFF_R2 = 134217728, OFF_R3 = 201326592,
                 OFF_R4 = 268435456, OFF_R5 = 335544320, OFF_W = 402653184,
                 OFF_ST = 403701760, NEED = 403702784;
    if (ws_size < NEED) return;

    unsigned short* R0   = (unsigned short*)ws;             // bf16 in / corr (reuse)
    unsigned short* R1   = (unsigned short*)(ws + OFF_R1);  // pre out bf16 (8 imgs)
    unsigned short* FC1O = (unsigned short*)(ws + OFF_R2);  // fc1 out bf16
    float* E0   = (float*)(ws + OFF_R2);                    // reuse after fc2
    float* E1   = (float*)(ws + OFF_R2 + 33554432);
    float* E2   = (float*)(ws + OFF_R2 + 50331648);
    float* E3   = (float*)(ws + OFF_R2 + 58720256);
    float* BT   = (float*)(ws + OFF_R2 + 60817408);
    float* FEAT = (float*)(ws + OFF_R3);                    // fc2 out f32
    float* F1B  = (float*)(ws + OFF_R4);                    // blurred f1 f32
    float* TMP  = (float*)(ws + OFF_R5);                    // blur intermediate
    unsigned short* Wpre = (unsigned short*)(ws + OFF_W);
    unsigned short* Wfc1 = Wpre + 64 * 576;
    unsigned short* Wfc2 = Wfc1 + 128 * 1152;
    unsigned short* We0  = Wfc2 + 64 * 1152;
    float* STATS = (float*)(ws + OFF_ST);

    const int T = 256;
    auto blk = [](size_t n) { return (unsigned)((n + 255) / 256); };

    // 1) inputs -> bf16 (concat along batch: 8 images of 64ch)
    cvt_f32_bf16<<<blk(4 * 64 * HW), T, 0, stream>>>(feat1, R0, 4 * 64 * HW);
    cvt_f32_bf16<<<blk(4 * 64 * HW), T, 0, stream>>>(feat2, R0 + 4 * 64 * HW, 4 * 64 * HW);

    // 2) pack weights into WMMA A-fragment order (bf16)
    pack_w<<<blk(64 * 576),   T, 0, stream>>>(w_pre, Wpre, 64, 64, 64);
    pack_w<<<blk(128 * 1152), T, 0, stream>>>(w_fc1, Wfc1, 128, 128, 128);
    pack_w<<<blk(64 * 1152),  T, 0, stream>>>(w_fc2, Wfc2, 64, 128, 128);
    pack_w<<<blk(32 * 1152),  T, 0, stream>>>(w_e0,  We0,  32, 113, 128);

    // 3) pre conv: (8,64)->(8,64), bf16 out
    conv3x3_wmma<64, 64, false, true><<<8 * HW / 16, 64, 0, stream>>>(
        R0, R0, 64, Wpre, nullptr, R1);

    // 4) fc1: concat(f1,f2) along channels -> 128ch, lrelu, bf16 out
    conv3x3_wmma<128, 128, true, true><<<4 * HW / 16, 64, 0, stream>>>(
        R1, R1 + 4 * 64 * HW, 64, Wfc1, nullptr, FC1O);

    // 5) fc2: 128->64, f32 out (feat)
    conv3x3_wmma<128, 64, false, false><<<4 * HW / 16, 64, 0, stream>>>(
        FC1O, FC1O, 128, Wfc2, FEAT, nullptr);

    // 6) blur13(f1, zero-pad): vertical then horizontal
    blur13_pass<true,  false, true,  false><<<blk(256 * HW), T, 0, stream>>>(R1, TMP, 256 * HW);
    blur13_pass<false, false, false, false><<<blk(256 * HW), T, 0, stream>>>(TMP, F1B, 256 * HW);

    // 7) corr tensor: feat -> ch 0..63 (bf16), zero ch 113..127
    feat_to_corr<<<blk(4 * 128 * HW), T, 0, stream>>>(FEAT, R0, 4 * 128 * HW);

    // 8) 49-shift local correlation -> corr ch 64..112
    localcorr_k<<<blk(4 * 49 * HW), T, 0, stream>>>(R1 + 4 * 64 * HW, F1B, R0, 4 * 49 * HW);

    // 9) e0: 128(113)->32 via WMMA, f32 out
    conv3x3_wmma<128, 32, false, false><<<4 * HW / 16, 64, 0, stream>>>(
        R0, R0, 128, We0, E0, nullptr);

    // 10) BN + lrelu (e0)
    zero_f32<<<1, 64, 0, stream>>>(STATS, 64);
    bn_stats<<<blk(4 * 32 * HW), T, 0, stream>>>(E0, STATS, 32, 4 * 32 * HW);
    bn_apply_lrelu<<<blk(4 * 32 * HW), T, 0, stream>>>(E0, STATS, 32, 4 * 32 * HW);

    // 11) e1: 32->16, dil 2, then BN+lrelu
    conv3x3_direct<<<blk(4 * 16 * HW), T, 0, stream>>>(E0, w_e1, E1, 32, 16, 2, 4 * 16 * HW);
    zero_f32<<<1, 64, 0, stream>>>(STATS, 32);
    bn_stats<<<blk(4 * 16 * HW), T, 0, stream>>>(E1, STATS, 16, 4 * 16 * HW);
    bn_apply_lrelu<<<blk(4 * 16 * HW), T, 0, stream>>>(E1, STATS, 16, 4 * 16 * HW);

    // 12) e2: 16->8, dil 4, then BN+lrelu
    conv3x3_direct<<<blk(4 * 8 * HW), T, 0, stream>>>(E1, w_e2, E2, 16, 8, 4, 4 * 8 * HW);
    zero_f32<<<1, 64, 0, stream>>>(STATS, 16);
    bn_stats<<<blk(4 * 8 * HW), T, 0, stream>>>(E2, STATS, 8, 4 * 8 * HW);
    bn_apply_lrelu<<<blk(4 * 8 * HW), T, 0, stream>>>(E2, STATS, 8, 4 * 8 * HW);

    // 13) e3: 8->2, dil 1, no BN/act
    conv3x3_direct<<<blk(4 * 2 * HW), T, 0, stream>>>(E2, w_e3, E3, 8, 2, 1, 4 * 2 * HW);

    // 14) blur13(edge) + clip + per-channel scale -> d_out
    blur13_pass<true,  true, false, false><<<blk(8 * HW), T, 0, stream>>>(E3, BT, 8 * HW);
    blur13_pass<false, true, false, true ><<<blk(8 * HW), T, 0, stream>>>(BT, (float*)d_out, 8 * HW);
}
